// SimpleLSTM_53249004536456
// MI455X (gfx1250) — matrix operand order repaired and checked by reading, not compiled, not run on hardware
//
#include <hip/hip_runtime.h>

// CDNA5 / gfx1250: wave32, WMMA f32_16x16x32_f16.
// Fused 2-layer LSTM (H=32) + Linear head. One workgroup = 2 waves = one
// 16-batch tile; wave w owns hidden half w (M-tiles {w,2+w,4+w,6+w}).
// M = gates (128), N = batch (16), K = hidden (32). h staging is
// double-buffered in LDS so only producer->consumer barriers are needed.

typedef __attribute__((ext_vector_type(16))) _Float16     v16h;
typedef __attribute__((ext_vector_type(8)))  float        v8f;
typedef __attribute__((ext_vector_type(4)))  float        v4f;
typedef __attribute__((ext_vector_type(4)))  unsigned int v4u;

union Frag16 {
  v16h h;
  v4u  u[2];
};

__device__ __forceinline__ float tanh_hw(float x) {
#if __has_builtin(__builtin_amdgcn_tanhf)
  return __builtin_amdgcn_tanhf(x);          // v_tanh_f32 (gfx1250 trans op)
#else
  float xc = fminf(fmaxf(x, -15.0f), 15.0f);
  float e  = __expf(2.0f * xc);
  return (e - 1.0f) * __builtin_amdgcn_rcpf(e + 1.0f);
#endif
}
__device__ __forceinline__ float sigm(float x) {
#if __has_builtin(__builtin_amdgcn_tanhf)
  return fmaf(0.5f, __builtin_amdgcn_tanhf(0.5f * x), 0.5f);
#else
  return __builtin_amdgcn_rcpf(1.0f + __expf(-x));
#endif
}

#define HDIM 32
#define GDIM 128   /* 4*H */

__global__ __launch_bounds__(64)
void lstm2_wmma_kernel(const float* __restrict__ x,
                       const float* __restrict__ Wih0,
                       const float* __restrict__ Whh0,
                       const float* __restrict__ bih0,
                       const float* __restrict__ bhh0,
                       const float* __restrict__ Wih1,
                       const float* __restrict__ Whh1,
                       const float* __restrict__ bih1,
                       const float* __restrict__ bhh1,
                       const float* __restrict__ Wfc,
                       const float* __restrict__ bfc,
                       float* __restrict__ out,
                       int S)
{
  // f16 weights, row-major [gate][k] (64B rows): WMMA A-fragment per lane is
  // two contiguous 16B chunks (ds_load_b128 x2).
  __shared__ __align__(16) _Float16 s_whh0[GDIM * HDIM];
  __shared__ __align__(16) _Float16 s_wih1[GDIM * HDIM];
  __shared__ __align__(16) _Float16 s_whh1[GDIM * HDIM];
  __shared__ __align__(16) float    s_w0[GDIM];   // Wih0[:,0]
  __shared__ __align__(16) float    s_b0[GDIM];   // bih0 + bhh0
  __shared__ __align__(16) float    s_b1[GDIM];   // bih1 + bhh1
  __shared__ __align__(16) float    s_wfc[HDIM];
  // double-buffered h staging (f16), row-major [batch][hidden]
  __shared__ __align__(16) _Float16 s_h0[2][16 * HDIM];
  __shared__ __align__(16) _Float16 s_h1[2][16 * HDIM];
  __shared__ __align__(16) float    s_red[2][16];

  const int tid   = threadIdx.x;
  const int ln    = tid & 31;
  const int wv    = tid >> 5;      // wave id: hidden half (0: h 0-15, 1: h 16-31)
  const int lq    = ln & 15;       // batch column within tile
  const int lh    = ln >> 4;       // wave half (A/B K-split, D M-split)
  const int bbase = blockIdx.x * 16;

  // ---- one-time: convert weights to f16 in LDS, fold biases ----
  for (int i = tid; i < GDIM * HDIM; i += 64) {
    s_whh0[i] = (_Float16)Whh0[i];
    s_wih1[i] = (_Float16)Wih1[i];
    s_whh1[i] = (_Float16)Whh1[i];
  }
  for (int i = tid; i < GDIM; i += 64) {
    s_w0[i] = Wih0[i];
    s_b0[i] = bih0[i] + bhh0[i];
    s_b1[i] = bih1[i] + bhh1[i];
  }
  for (int i = tid; i < 16 * HDIM; i += 64) {
    s_h0[0][i] = (_Float16)0.0f;  s_h0[1][i] = (_Float16)0.0f;
    s_h1[0][i] = (_Float16)0.0f;  s_h1[1][i] = (_Float16)0.0f;
  }
  if (tid < HDIM) s_wfc[tid] = Wfc[tid];
  __syncthreads();

  // state regs: index r -> hidden = 16*wv + 8*lh + r, batch = lq
  float c0[8], c1[8], h1r[8];
  #pragma unroll
  for (int i = 0; i < 8; ++i) { c0[i] = 0.f; c1[i] = 0.f; h1r[i] = 0.f; }

  const float* xrow = x + (size_t)(bbase + lq) * S;  // D_IN == 1
  const int hoff = lq * HDIM + lh * 16;  // B-fragment base in h staging
  const int aoff = lh * 16;              // A-fragment K sub-offset
  const int soff = lq * HDIM + 16 * wv + 8 * lh;  // this wave's h store slot

  int rd = 0;
  for (int t = 0; t < S; ++t) {
    const int wr = rd ^ 1;
    float xv = xrow[t];
    if ((t & 63) == 0) __builtin_prefetch(&xrow[t + 64], 0, 1);

    // ============ Layer 0: g0 = Whh0 · h0_prevᵀ + x·w0 + b0 (my 4 tiles) ===
    Frag16 bh;  // B fragment: h0_prevᵀ (full K=32)
    {
      const _Float16* p = &s_h0[rd][hoff];
      bh.u[0] = *(const v4u*)p;
      bh.u[1] = *(const v4u*)(p + 8);
    }

    v8f acc[4];
    #pragma unroll
    for (int mm = 0; mm < 4; ++mm) {          // mm: 0=i 1=f 2=g 3=o
      const int m  = 2 * mm + wv;             // actual M-tile
      const int gb = 16 * m + 8 * lh;
      v4f b_lo = *(const v4f*)&s_b0[gb];
      v4f b_hi = *(const v4f*)&s_b0[gb + 4];
      v4f w_lo = *(const v4f*)&s_w0[gb];
      v4f w_hi = *(const v4f*)&s_w0[gb + 4];
      v8f c;
      #pragma unroll
      for (int r = 0; r < 4; ++r) {
        c[r]     = b_lo[r] + xv * w_lo[r];    // rank-1 input + bias
        c[r + 4] = b_hi[r] + xv * w_hi[r];
      }
      Frag16 a;
      const _Float16* wp = &s_whh0[(16 * m + lq) * HDIM + aoff];
      a.u[0] = *(const v4u*)wp;
      a.u[1] = *(const v4u*)(wp + 8);
      acc[mm] = __builtin_amdgcn_wmma_f32_16x16x32_f16(false, a.h, false, bh.h,
                                                       (short)0, c, false, false);
    }

    {
      union { _Float16 hf[8]; v4u u; } pk;
      #pragma unroll
      for (int r = 0; r < 8; ++r) {
        float iv = sigm(acc[0][r]);
        float fv = sigm(acc[1][r]);
        float gv = tanh_hw(acc[2][r]);
        float ov = sigm(acc[3][r]);
        float cn = fv * c0[r] + iv * gv;
        c0[r] = cn;
        pk.hf[r] = (_Float16)(ov * tanh_hw(cn));
      }
      *(v4u*)&s_h0[wr][soff] = pk.u;          // one ds_store_b128
    }

    __syncthreads();  // h0_t (both hidden halves) visible

    // ====== Layer 1: g1 = Wih1·h0_tᵀ + Whh1·h1_prevᵀ + b1 (my 4 tiles) =====
    Frag16 bx, bhp;
    {
      const _Float16* p0 = &s_h0[wr][hoff];   // fresh layer-0 hidden
      bx.u[0]  = *(const v4u*)p0;
      bx.u[1]  = *(const v4u*)(p0 + 8);
      const _Float16* p1 = &s_h1[rd][hoff];   // layer-1 previous hidden
      bhp.u[0] = *(const v4u*)p1;
      bhp.u[1] = *(const v4u*)(p1 + 8);
    }

    #pragma unroll
    for (int mm = 0; mm < 4; ++mm) {
      const int m  = 2 * mm + wv;
      const int gb = 16 * m + 8 * lh;
      v4f b_lo = *(const v4f*)&s_b1[gb];
      v4f b_hi = *(const v4f*)&s_b1[gb + 4];
      v8f c;
      #pragma unroll
      for (int r = 0; r < 4; ++r) { c[r] = b_lo[r]; c[r + 4] = b_hi[r]; }
      Frag16 a;
      const _Float16* wp = &s_wih1[(16 * m + lq) * HDIM + aoff];
      a.u[0] = *(const v4u*)wp;
      a.u[1] = *(const v4u*)(wp + 8);
      c = __builtin_amdgcn_wmma_f32_16x16x32_f16(false, a.h, false, bx.h,
                                                 (short)0, c, false, false);
      const _Float16* wp2 = &s_whh1[(16 * m + lq) * HDIM + aoff];
      a.u[0] = *(const v4u*)wp2;
      a.u[1] = *(const v4u*)(wp2 + 8);
      acc[mm] = __builtin_amdgcn_wmma_f32_16x16x32_f16(false, a.h, false, bhp.h,
                                                       (short)0, c, false, false);
    }

    {
      union { _Float16 hf[8]; v4u u; } pk;
      #pragma unroll
      for (int r = 0; r < 8; ++r) {
        float iv = sigm(acc[0][r]);
        float fv = sigm(acc[1][r]);
        float gv = tanh_hw(acc[2][r]);
        float ov = sigm(acc[3][r]);
        float cn = fv * c1[r] + iv * gv;
        c1[r] = cn;
        float hv = ov * tanh_hw(cn);
        h1r[r] = hv;                           // keep f32 for the head
        pk.hf[r] = (_Float16)hv;
      }
      *(v4u*)&s_h1[wr][soff] = pk.u;
    }

    __syncthreads();  // h1_t visible for next step
    rd = wr;
  }

  // ---- head: out[b] = h1_T · Wfc + bfc (reduce over lh halves + waves) ----
  float res = 0.f;
  #pragma unroll
  for (int r = 0; r < 8; ++r)
    res += h1r[r] * s_wfc[16 * wv + 8 * lh + r];
  res += __shfl_xor(res, 16, 32);             // combine the two wave halves
  if (lh == 0) s_red[wv][lq] = res;
  __syncthreads();
  if (tid < 16) out[bbase + tid] = s_red[0][tid] + s_red[1][tid] + bfc[0];
}

extern "C" void kernel_launch(void* const* d_in, const int* in_sizes, int n_in,
                              void* d_out, int out_size, void* d_ws, size_t ws_size,
                              hipStream_t stream) {
  const float* x    = (const float*)d_in[0];
  const float* Wih0 = (const float*)d_in[1];
  const float* Whh0 = (const float*)d_in[2];
  const float* bih0 = (const float*)d_in[3];
  const float* bhh0 = (const float*)d_in[4];
  const float* Wih1 = (const float*)d_in[5];
  const float* Whh1 = (const float*)d_in[6];
  const float* bih1 = (const float*)d_in[7];
  const float* bhh1 = (const float*)d_in[8];
  const float* Wfc  = (const float*)d_in[9];
  const float* bfc  = (const float*)d_in[10];
  float* out = (float*)d_out;

  const int S = 512;
  const int B = in_sizes[0] / S;   // D_IN == 1 -> x has B*S elements
  dim3 grid(B / 16), block(64);    // 2 waves per 16-batch tile
  hipLaunchKernelGGL(lstm2_wmma_kernel, grid, block, 0, stream,
                     x, Wih0, Whh0, bih0, bhh0,
                     Wih1, Whh1, bih1, bhh1, Wfc, bfc, out, S);
}
